// VanillaRNN_64931315581254
// MI455X (gfx1250) — compile-verified, matrix operand order checked by reading, not run on hardware
//
#include <hip/hip_runtime.h>
#include <hip/hip_bf16.h>
#include <math.h>
#include <stdint.h>

// ---------------------------------------------------------------------------
// VanillaRNN on MI455X (gfx1250, wave32, WMMA):
//   xp = x @ W_hx^T + b_hx            (one big bf16 WMMA GEMM, stored [T,B,H])
//   for t: h = tanh(xp_t + h @ W_hh^T + (b_hh+b_h))   (512 small WMMA GEMMs)
//   o = h @ W_ph^T + b_ph + b_o ; log_softmax(o)
// Matmuls use v_wmma_f32_16x16x32_bf16 (f32 accumulate).  The recurrence step
// stages W_hh tiles into LDS with global_load_async_to_lds_b128 (ASYNCcnt,
// double-buffered) so all 8 waves of a block share one copy of the B tile.
// ---------------------------------------------------------------------------

typedef __attribute__((ext_vector_type(16))) __bf16 v16bf;
typedef __attribute__((ext_vector_type(8)))  __bf16 v8bf;
typedef __attribute__((ext_vector_type(8)))  float  v8f;

#define B_ 256
#define T_ 512
#define D_ 256
#define H_ 1024
#define O_ 128
#define KC 64   // K chunk staged into LDS per double-buffer slot

// ---------------------------------------------------------------------------
// Load a 16x32 bf16 fragment (A layout, ISA 7.12.2) from a row-major matrix.
// Lane L (0..15): row = L,    K = 0..7  (elems 0..7),  K = 16..23 (elems 8..15)
// Lane L (16..31): row = L-16, K = 8..15 (elems 0..7), K = 24..31 (elems 8..15)
// The same pattern loads the B fragment for C = X * W^T, because B's column n
// is W's row n (both are contiguous runs of K in row-major storage).
// Works for global or LDS source pointers (generic addressing).
// ---------------------------------------------------------------------------
__device__ __forceinline__ v16bf load_frag_rm(const __bf16* tile, int ld) {
    const int lane = threadIdx.x & 31;
    const int r    = lane & 15;
    const int kb   = (lane >> 4) << 3;          // 0 or 8
    const v8bf* p  = (const v8bf*)(tile + (size_t)r * ld + kb);
    v8bf lo = p[0];                             // K = kb + 0..7
    v8bf hi = p[2];                             // K = kb + 16..23
    v16bf f;
#pragma unroll
    for (int i = 0; i < 8; ++i) { f[i] = lo[i]; f[i + 8] = hi[i]; }
    return f;
}

__device__ __forceinline__ v8f wmma_bf16(v16bf a, v16bf b, v8f c) {
    return __builtin_amdgcn_wmma_f32_16x16x32_bf16(false, a, false, b,
                                                   (short)0, c, false, false);
}

// ---------------------------------------------------------------------------
// Elementwise helpers
// ---------------------------------------------------------------------------
__global__ void f32_to_bf16_kernel(const float* __restrict__ in,
                                   __bf16* __restrict__ out, int n) {
    int i = blockIdx.x * blockDim.x + threadIdx.x;
    if (i < n) out[i] = (__bf16)in[i];
}

__global__ void zero_bf16_kernel(__bf16* __restrict__ out, int n) {
    int i = blockIdx.x * blockDim.x + threadIdx.x;
    if (i < n) out[i] = (__bf16)0.0f;
}

__global__ void bias_combine_kernel(const float* __restrict__ a,
                                    const float* __restrict__ b,
                                    float* __restrict__ out, int n) {
    int i = blockIdx.x * blockDim.x + threadIdx.x;
    if (i < n) out[i] = a[i] + b[i];
}

// ---------------------------------------------------------------------------
// xp GEMM:  XP[t,b,h] = sum_k X[b,t,k] * Whx[h,k] + bhx[h]
// M = B*T rows of X (row m -> b = m/T, t = m%T), N = H, K = D.
// Block = 256 thr = 8 waves; wave computes a 16(M) x 64(N) strip.
// Grid = (M/128, H/64) = (1024, 16).
// ---------------------------------------------------------------------------
__global__ void __launch_bounds__(256)
xp_gemm_kernel(const __bf16* __restrict__ X, const __bf16* __restrict__ Whx,
               const float* __restrict__ bhx, float* __restrict__ XP) {
    const int wave = threadIdx.x >> 5;
    const int lane = threadIdx.x & 31;
    const int row0 = (blockIdx.x * 8 + wave) * 16;
    const int col0 = blockIdx.y * 64;

    v8f acc[4] = {};
    for (int k = 0; k < D_; k += 32) {
        v16bf a = load_frag_rm(X + (size_t)row0 * D_ + k, D_);
#pragma unroll
        for (int j = 0; j < 4; ++j) {
            v16bf b = load_frag_rm(Whx + (size_t)(col0 + j * 16) * D_ + k, D_);
            acc[j] = wmma_bf16(a, b, acc[j]);
        }
    }

    const int c     = lane & 15;
    const int rbase = (lane >> 4) << 3;
#pragma unroll
    for (int j = 0; j < 4; ++j) {
        const int col = col0 + j * 16 + c;
        const float bias = bhx[col];
#pragma unroll
        for (int v = 0; v < 8; ++v) {
            const int m  = row0 + rbase + v;
            const int t  = m & (T_ - 1);         // T = 512 (pow2)
            const int bb = m >> 9;
            XP[((size_t)t * B_ + bb) * H_ + col] = acc[j][v] + bias;
        }
    }
}

// ---------------------------------------------------------------------------
// One recurrence step: Hout = tanh(XPt + Hin @ Whh^T + bias), bf16 out.
// M = B = 256, N = H = 1024, K = H = 1024.  Wave = 16(M) x 64(N) strip;
// block = 8 waves = 128(M) x 64(N).  Grid = (2, 16), block = 256.
//
// The 64 W_hh rows needed by the whole block are staged into LDS in KC=64
// K-chunks with global_load_async_to_lds_b128 (ASYNCcnt), double-buffered so
// chunk kc+1's DMA overlaps chunk kc's 8 WMMAs.  B fragments then come from
// LDS (ds_load_b128), cutting W_hh fetch traffic 8x per block.
// ---------------------------------------------------------------------------
__global__ void __launch_bounds__(256)
rnn_step_kernel(const __bf16* __restrict__ Hin, const float* __restrict__ XPt,
                const __bf16* __restrict__ Whh, const float* __restrict__ bias,
                __bf16* __restrict__ Hout) {
    __shared__ __bf16 wtile[2][64 * KC];        // 2 x 8 KB double buffer
    const int tid  = threadIdx.x;
    const int wave = tid >> 5;
    const int lane = tid & 31;
    const int row0 = (blockIdx.x * 8 + wave) * 16;
    const int col0 = blockIdx.y * 64;

    // Stage W_hh rows [col0, col0+64) x K [kb, kb+KC) into wtile[buf].
    // 8 KB = 512 x 16B chunks; 256 threads issue 2 async b128 copies each.
    auto stage = [&](int buf, int kb) {
#pragma unroll
        for (int i = 0; i < 2; ++i) {
            const int c  = tid + i * 256;                 // chunk 0..511
            const int r  = c >> 3;                        // row in tile
            const int kk = (c & 7) * 8;                   // elem offset in row
            const __bf16* g = Whh + (size_t)(col0 + r) * H_ + kb + kk;
            // LDS byte address = low 32 bits of the generic pointer
            // (flat-aperture truncation rule, ISA 10.2).
            uint32_t lds = (uint32_t)(uintptr_t)(&wtile[buf][r * KC + kk]);
            uint64_t ga  = (uint64_t)(uintptr_t)g;
            asm volatile("global_load_async_to_lds_b128 %0, %1, off"
                         :: "v"(lds), "v"(ga)
                         : "memory");
        }
    };

    stage(0, 0);

    v8f acc[4] = {};
    for (int kc = 0; kc < H_ / KC; ++kc) {
        // own async copies landed ...
        asm volatile("s_wait_asynccnt 0x0" ::: "memory");
        // ... and everyone else's too
        __syncthreads();
        if (kc + 1 < H_ / KC) stage((kc + 1) & 1, (kc + 1) * KC);

        const __bf16* wl = &wtile[kc & 1][0];
#pragma unroll
        for (int kk = 0; kk < KC; kk += 32) {
            v16bf a = load_frag_rm(Hin + (size_t)row0 * H_ + kc * KC + kk, H_);
#pragma unroll
            for (int j = 0; j < 4; ++j) {
                v16bf b = load_frag_rm(wl + (j * 16) * KC + kk, KC);
                acc[j] = wmma_bf16(a, b, acc[j]);
            }
        }
        // all waves done reading wtile[kc&1] before it is restaged at kc+2
        __syncthreads();
    }

    const int c     = lane & 15;
    const int rbase = (lane >> 4) << 3;
#pragma unroll
    for (int j = 0; j < 4; ++j) {
        const int col = col0 + j * 16 + c;
        const float bs = bias[col];
#pragma unroll
        for (int v = 0; v < 8; ++v) {
            const int m = row0 + rbase + v;
            const float val = acc[j][v] + XPt[(size_t)m * H_ + col] + bs;
            Hout[(size_t)m * H_ + col] = (__bf16)tanhf(val);
        }
    }
}

// ---------------------------------------------------------------------------
// Output head: logits = Hf @ Wph^T + (bph + bo).  M=256, N=128, K=1024.
// Wave = one 16x16 tile. Grid = (2, 8), block = 256.
// ---------------------------------------------------------------------------
__global__ void __launch_bounds__(256)
out_gemm_kernel(const __bf16* __restrict__ Hf, const __bf16* __restrict__ Wph,
                const float* __restrict__ bph, const float* __restrict__ bo,
                float* __restrict__ logits) {
    const int wave = threadIdx.x >> 5;
    const int lane = threadIdx.x & 31;
    const int row0 = (blockIdx.x * 8 + wave) * 16;
    const int col0 = blockIdx.y * 16;

    v8f acc = {};
    for (int k = 0; k < H_; k += 32) {
        v16bf a = load_frag_rm(Hf  + (size_t)row0 * H_ + k, H_);
        v16bf b = load_frag_rm(Wph + (size_t)col0 * H_ + k, H_);
        acc = wmma_bf16(a, b, acc);
    }

    const int c     = lane & 15;
    const int rbase = (lane >> 4) << 3;
    const int col   = col0 + c;
    const float bs  = bph[col] + bo[col];
#pragma unroll
    for (int v = 0; v < 8; ++v) {
        const int m = row0 + rbase + v;
        logits[(size_t)m * O_ + col] = acc[v] + bs;
    }
}

// ---------------------------------------------------------------------------
// Row-wise log_softmax over O=128.  One block (4 wave32) per row.
// ---------------------------------------------------------------------------
__global__ void __launch_bounds__(128)
log_softmax_kernel(const float* __restrict__ logits, float* __restrict__ out) {
    const int row  = blockIdx.x;
    const int tid  = threadIdx.x;
    const int lane = tid & 31;
    const int wid  = tid >> 5;
    __shared__ float rmax[4];
    __shared__ float rsum[4];

    float v = logits[(size_t)row * O_ + tid];

    float m = v;
#pragma unroll
    for (int off = 16; off > 0; off >>= 1)
        m = fmaxf(m, __shfl_xor(m, off, 32));
    if (lane == 0) rmax[wid] = m;
    __syncthreads();
    m = fmaxf(fmaxf(rmax[0], rmax[1]), fmaxf(rmax[2], rmax[3]));

    float e = expf(v - m);
    float s = e;
#pragma unroll
    for (int off = 16; off > 0; off >>= 1)
        s += __shfl_xor(s, off, 32);
    if (lane == 0) rsum[wid] = s;
    __syncthreads();
    s = rsum[0] + rsum[1] + rsum[2] + rsum[3];

    out[(size_t)row * O_ + tid] = (v - m) - logf(s);
}

// ---------------------------------------------------------------------------
// Host-side orchestration
// ---------------------------------------------------------------------------
extern "C" void kernel_launch(void* const* d_in, const int* in_sizes, int n_in,
                              void* d_out, int out_size, void* d_ws, size_t ws_size,
                              hipStream_t stream) {
    const float* x    = (const float*)d_in[0];   // [B,T,D]
    const float* Whx  = (const float*)d_in[1];   // [H,D]
    const float* bhx  = (const float*)d_in[2];   // [H]
    const float* Whh  = (const float*)d_in[3];   // [H,H]
    const float* bhh  = (const float*)d_in[4];   // [H]
    const float* bh   = (const float*)d_in[5];   // [H]
    const float* Wph  = (const float*)d_in[6];   // [O,H]
    const float* bph  = (const float*)d_in[7];   // [O]
    const float* bo   = (const float*)d_in[8];   // [O]
    float* out        = (float*)d_out;           // [B,O]

    // Workspace carve-up (256-byte aligned slabs)
    char* ws = (char*)d_ws;
    size_t off = 0;
    auto carve = [&](size_t bytes) -> char* {
        char* p = ws + off;
        off += (bytes + 255) & ~(size_t)255;
        return p;
    };
    __bf16* x_b    = (__bf16*)carve((size_t)B_ * T_ * D_ * 2);  // 67 MB
    __bf16* Whx_b  = (__bf16*)carve((size_t)H_ * D_ * 2);
    __bf16* Whh_b  = (__bf16*)carve((size_t)H_ * H_ * 2);
    __bf16* Wph_b  = (__bf16*)carve((size_t)O_ * H_ * 2);
    float*  XP     = (float*) carve((size_t)B_ * T_ * H_ * 4);  // 536 MB, [T,B,H]
    __bf16* h_a    = (__bf16*)carve((size_t)B_ * H_ * 2);
    __bf16* h_b    = (__bf16*)carve((size_t)B_ * H_ * 2);
    float*  biasc  = (float*) carve((size_t)H_ * 4);
    float*  logits = (float*) carve((size_t)B_ * O_ * 4);
    (void)ws_size; (void)in_sizes; (void)n_in; (void)out_size;

    // 1) fp32 -> bf16 conversions
    {
        int n = B_ * T_ * D_;
        f32_to_bf16_kernel<<<(n + 255) / 256, 256, 0, stream>>>(x, x_b, n);
        n = H_ * D_;
        f32_to_bf16_kernel<<<(n + 255) / 256, 256, 0, stream>>>(Whx, Whx_b, n);
        n = H_ * H_;
        f32_to_bf16_kernel<<<(n + 255) / 256, 256, 0, stream>>>(Whh, Whh_b, n);
        n = O_ * H_;
        f32_to_bf16_kernel<<<(n + 255) / 256, 256, 0, stream>>>(Wph, Wph_b, n);
    }
    // 2) bias = b_hh + b_h ; h0 = 0
    bias_combine_kernel<<<(H_ + 255) / 256, 256, 0, stream>>>(bhh, bh, biasc, H_);
    {
        int n = B_ * H_;
        zero_bf16_kernel<<<(n + 255) / 256, 256, 0, stream>>>(h_a, n);
    }

    // 3) xp = x @ Whx^T + bhx, stored [T,B,H]
    {
        dim3 grid((B_ * T_) / 128, H_ / 64);
        xp_gemm_kernel<<<grid, 256, 0, stream>>>(x_b, Whx_b, bhx, XP);
    }

    // 4) Recurrence: 512 sequential WMMA steps (ping-pong bf16 h buffers)
    {
        dim3 grid(B_ / 128, H_ / 64);
        __bf16* cur = h_a;
        __bf16* nxt = h_b;
        for (int t = 0; t < T_; ++t) {
            rnn_step_kernel<<<grid, 256, 0, stream>>>(
                cur, XP + (size_t)t * B_ * H_, Whh_b, biasc, nxt);
            __bf16* tmp = cur; cur = nxt; nxt = tmp;
        }
        // T even -> final h lives in h_a (== cur after the loop)
        // 5) output head
        dim3 grid2(B_ / 128, O_ / 16);
        out_gemm_kernel<<<grid2, 256, 0, stream>>>(cur, Wph_b, bph, bo, logits);
    }

    // 6) log_softmax rows
    log_softmax_kernel<<<B_, 128, 0, stream>>>(logits, out);
}
